// CustomGPT1Model_41317585388103
// MI455X (gfx1250) — matrix operand order repaired and checked by reading, not compiled, not run on hardware
//
#include <hip/hip_runtime.h>

typedef __attribute__((ext_vector_type(16))) __bf16 v16bf;
typedef __attribute__((ext_vector_type(8)))  float  v8f;
typedef unsigned int u32x4 __attribute__((ext_vector_type(4)));
typedef int          i32x8 __attribute__((ext_vector_type(8)));
typedef int          i32x4 __attribute__((ext_vector_type(4)));

union FragBF {
  v16bf v;
  unsigned short u[16];
  unsigned int   w[8];
};

__device__ __forceinline__ unsigned short f2bfu(float f) {
  union { float f; unsigned int u; } c; c.f = f;
  unsigned int u = c.u + 0x7FFFu + ((c.u >> 16) & 1u);
  return (unsigned short)(u >> 16);
}
__device__ __forceinline__ float bfu2f(unsigned short s) {
  union { float f; unsigned int u; } c; c.u = ((unsigned int)s) << 16; return c.f;
}
__device__ __forceinline__ unsigned int pack2bf(float lo, float hi) {
  return (unsigned int)f2bfu(lo) | ((unsigned int)f2bfu(hi) << 16);
}

// ---------------------------------------------------------------------------
// Tensor Data Mover helpers (gfx1250): async 2D bf16 tile load global -> LDS.
// D# layout per CDNA5 ISA. TENSORcnt tracks completion.
// ---------------------------------------------------------------------------
#if defined(__gfx1250__)
#define TDM_WAIT(n)  __builtin_amdgcn_s_wait_tensorcnt(n)
#define LDS_DRAIN()  asm volatile("s_wait_dscnt 0" ::: "memory")
#else
#define TDM_WAIT(n)
#define LDS_DRAIN()
#endif

__device__ __forceinline__ void tdm_load_2d_bf16(unsigned int lds_off,
                                                 const void* gaddr,
                                                 unsigned int stride0,   // elems
                                                 unsigned int tile_d0,   // elems
                                                 unsigned int tile_d1) { // rows
#if defined(__gfx1250__)
  unsigned long long ga = (unsigned long long)gaddr;
  u32x4 g0;
  g0.x = 1u;                                   // count=1, no gather
  g0.y = lds_off;                              // LDS byte address
  g0.z = (unsigned int)ga;                     // global addr [31:0]
  g0.w = (unsigned int)((ga >> 32) & 0x01FFFFFFu) | 0x80000000u; // [56:32] | type=2
  i32x8 g1;
  g1[0] = (int)(1u << 16);                                  // data_size=2B
  g1[1] = (int)((tile_d0 & 0xFFFFu) << 16);                 // tensor_dim0 lo16
  g1[2] = (int)(((tile_d0 >> 16) & 0xFFFFu) |
                ((tile_d1 & 0xFFFFu) << 16));               // dim0 hi | dim1 lo
  g1[3] = (int)(((tile_d1 >> 16) & 0xFFFFu) |
                ((tile_d0 & 0xFFFFu) << 16));               // dim1 hi | tile_dim0
  g1[4] = (int)(tile_d1 & 0xFFFFu);                         // tile_dim1, tile_dim2=0
  g1[5] = (int)stride0;                                     // tensor_dim0_stride lo
  g1[6] = 0;
  g1[7] = 0;
  i32x4 z4 = {0, 0, 0, 0};
#if __clang_major__ >= 23
  i32x8 z8 = {0, 0, 0, 0, 0, 0, 0, 0};
  __builtin_amdgcn_tensor_load_to_lds(g0, g1, z4, z4, z8, 0);
#else
  __builtin_amdgcn_tensor_load_to_lds(g0, g1, z4, z4, 0);
#endif
#endif
}

// ---------------------------------------------------------------------------
// Embedding: x[b,s,d] = tok[id]+pos[s]+attr[ci[b]]
// ---------------------------------------------------------------------------
__global__ void k_embed(const int* __restrict__ ids, const int* __restrict__ cidx,
                        const float* __restrict__ tok, const float* __restrict__ pos,
                        const float* __restrict__ attr, float* __restrict__ x) {
  int i = blockIdx.x * blockDim.x + threadIdx.x;   // B*S*D = 2^24 threads
  int d = i & 511;
  int s = (i >> 9) & 2047;
  int b = i >> 20;
  int id = ids[b * 2048 + s];
  int ai = cidx[b];
  x[i] = tok[id * 512 + d] + pos[s * 512 + d] + attr[ai * 512 + d];
}

// ---------------------------------------------------------------------------
// LayerNorm (D=512), one wave32 per row, bf16 output
// ---------------------------------------------------------------------------
__global__ __launch_bounds__(256) void k_layernorm(
    const float* __restrict__ x, const float* __restrict__ w,
    const float* __restrict__ bb, unsigned short* __restrict__ out, int rows) {
  int wid  = blockIdx.x * (blockDim.x >> 5) + (threadIdx.x >> 5);
  int lane = threadIdx.x & 31;
  if (wid >= rows) return;
  const float* xr = x + (size_t)wid * 512;
  float v[16];
  float s = 0.f;
#pragma unroll
  for (int i = 0; i < 16; ++i) { v[i] = xr[lane + i * 32]; s += v[i]; }
#pragma unroll
  for (int off = 16; off > 0; off >>= 1) s += __shfl_xor(s, off);
  float mu = s * (1.0f / 512.0f);
  float var = 0.f;
#pragma unroll
  for (int i = 0; i < 16; ++i) { float d = v[i] - mu; var += d * d; }
#pragma unroll
  for (int off = 16; off > 0; off >>= 1) var += __shfl_xor(var, off);
  float rstd = rsqrtf(var * (1.0f / 512.0f) + 1e-5f);
#pragma unroll
  for (int i = 0; i < 16; ++i) {
    int d = lane + i * 32;
    out[(size_t)wid * 512 + d] = f2bfu((v[i] - mu) * rstd * w[d] + bb[d]);
  }
}

// ---------------------------------------------------------------------------
// Fused attention: one wave per (batch, 16-row tile). Scores kept in LDS.
// All tile staging is done by the Tensor Data Mover with double buffering.
//   attn[b,m,:] = softmax(h h^T * scale + mask) h + x
// LDS map (dynamic, base 0):
//   q   @      0  16 KB   (16x512 bf16)
//   kt0 @  16384  16 KB
//   kt1 @  32768  16 KB
//   vt0 @  49152   8 KB   (32x128 bf16)
//   vt1 @  57344   8 KB
//   sc  @  65536  64 KB   (16x2048 bf16 scores)
//   st  @ 131072  128 B   (row max / inv-sum)
// ---------------------------------------------------------------------------
__global__ __launch_bounds__(32) void k_attention(
    const unsigned short* __restrict__ h, const float* __restrict__ x,
    const float* __restrict__ mask, float* __restrict__ attn) {
  constexpr int S = 2048, D = 512;
  constexpr unsigned OFF_Q = 0, OFF_K0 = 16384, OFF_K1 = 32768;
  constexpr unsigned OFF_V0 = 49152, OFF_V1 = 57344;
  constexpr unsigned OFF_SC = 65536, OFF_ST = 131072;
  extern __shared__ char smem[];
  unsigned short* q   = (unsigned short*)(smem + OFF_Q);
  unsigned short* ktA = (unsigned short*)(smem + OFF_K0);
  unsigned short* ktB = (unsigned short*)(smem + OFF_K1);
  unsigned short* vtA = (unsigned short*)(smem + OFF_V0);
  unsigned short* vtB = (unsigned short*)(smem + OFF_V1);
  unsigned short* sc  = (unsigned short*)(smem + OFF_SC);
  float*          st  = (float*)(smem + OFF_ST);

  int lane = threadIdx.x;
  int b  = blockIdx.x >> 7;          // S/16 = 128 tiles per batch
  int mt = blockIdx.x & 127;
  int m0 = mt * 16;
  const unsigned short* hb = h + (size_t)b * S * D;

  // TDM: stage Q tile and first K tile
  tdm_load_2d_bf16(OFF_Q, hb + (size_t)m0 * D, D, 512, 16);
  tdm_load_2d_bf16(OFF_K0, hb, D, 512, 16);
  TDM_WAIT(0);

  int half = lane >> 4;
  int r15  = lane & 15;
  float rm[8], rs[8];
#pragma unroll
  for (int r = 0; r < 8; ++r) { rm[r] = -3.0e38f; rs[r] = 0.f; }

  const float scale = 0.044194173824159216f;  // 1/sqrt(512)

  // ---- Pass A: scores + online row max / sumexp -------------------------
  for (int t = 0; t < S / 16; ++t) {
    if (t + 1 < S / 16) {  // prefetch next K tile into the other buffer
      LDS_DRAIN();
      tdm_load_2d_bf16(((t + 1) & 1) ? OFF_K1 : OFF_K0,
                       hb + (size_t)(t + 1) * 16 * D, D, 512, 16);
      TDM_WAIT(1);
    } else {
      TDM_WAIT(0);
    }
    const unsigned short* kt = (t & 1) ? ktB : ktA;

    v8f acc = {};
#pragma unroll
    for (int kc = 0; kc < 16; ++kc) {
      FragBF a, bf;
#pragma unroll
      for (int v2 = 0; v2 < 8; ++v2) {
        int ka = kc * 32 + (v2 >> 2) * 16 + (v2 & 3) * 2 + half * 8;  // A layout
        a.w[v2] = *(const unsigned int*)&q[r15 * 512 + ka];
        int kb = kc * 32 + (v2 >> 2) * 16 + (v2 & 3) * 2 + half * 8;  // B layout
        bf.w[v2] = *(const unsigned int*)&kt[r15 * 512 + kb];
      }
      acc = __builtin_amdgcn_wmma_f32_16x16x32_bf16(false, a.v, false, bf.v,
                                                    (short)0, acc, false, false);
    }
    int col = t * 16 + r15;
    float madd = (1.0f - mask[(size_t)b * S + col]) * -1.0e9f;
#pragma unroll
    for (int r = 0; r < 8; ++r) {
      float sv = acc[r] * scale + madd;
      float tm = sv;                       // row max within 16-lane group
      tm = fmaxf(tm, __shfl_xor(tm, 1));
      tm = fmaxf(tm, __shfl_xor(tm, 2));
      tm = fmaxf(tm, __shfl_xor(tm, 4));
      tm = fmaxf(tm, __shfl_xor(tm, 8));
      float mnew = fmaxf(rm[r], tm);
      float pe = __expf(sv - mnew);
      float ts = pe;
      ts += __shfl_xor(ts, 1); ts += __shfl_xor(ts, 2);
      ts += __shfl_xor(ts, 4); ts += __shfl_xor(ts, 8);
      rs[r] = rs[r] * __expf(rm[r] - mnew) + ts;
      rm[r] = mnew;
      sc[(r + 8 * half) * 2048 + col] = f2bfu(sv);
    }
  }

  if (r15 == 0) {
#pragma unroll
    for (int r = 0; r < 8; ++r) {
      st[r + 8 * half] = rm[r];
      st[16 + r + 8 * half] = 1.0f / rs[r];
    }
  }
  float rmx = st[r15];
  float rin = st[16 + r15];

  // ---- Pass B: O = P V, 128-wide D chunks, TDM-staged V tiles -----------
  for (int dc = 0; dc < 4; ++dc) {
    int d0 = dc * 128;
    LDS_DRAIN();
    tdm_load_2d_bf16(OFF_V0, hb + d0, D, 128, 32);  // first V tile (t0=0)

    v8f o[8];
#pragma unroll
    for (int nt = 0; nt < 8; ++nt) { v8f z = {}; o[nt] = z; }

    for (int i = 0; i < S / 32; ++i) {
      int t0 = i * 32;
      if (i + 1 < S / 32) {
        LDS_DRAIN();
        tdm_load_2d_bf16(((i + 1) & 1) ? OFF_V1 : OFF_V0,
                         hb + (size_t)(t0 + 32) * D + d0, D, 128, 32);
        TDM_WAIT(1);
      } else {
        TDM_WAIT(0);
      }
      const unsigned short* vt = (i & 1) ? vtB : vtA;

      FragBF a;
#pragma unroll
      for (int v2 = 0; v2 < 8; ++v2) {
        int ka = t0 + (v2 >> 2) * 16 + (v2 & 3) * 2 + half * 8;
        float s0 = bfu2f(sc[r15 * 2048 + ka]);
        float s1 = bfu2f(sc[r15 * 2048 + ka + 1]);
        a.w[v2] = pack2bf(__expf(s0 - rmx) * rin, __expf(s1 - rmx) * rin);
      }
#pragma unroll
      for (int nt = 0; nt < 8; ++nt) {
        FragBF bf;
#pragma unroll
        for (int v2 = 0; v2 < 8; ++v2) {
          int kb = (v2 >> 2) * 16 + (v2 & 3) * 2 + half * 8;
          bf.u[2 * v2]     = vt[kb * 128 + nt * 16 + r15];
          bf.u[2 * v2 + 1] = vt[(kb + 1) * 128 + nt * 16 + r15];
        }
        o[nt] = __builtin_amdgcn_wmma_f32_16x16x32_bf16(false, a.v, false, bf.v,
                                                        (short)0, o[nt], false, false);
      }
    }
    // epilogue: + residual x
#pragma unroll
    for (int nt = 0; nt < 8; ++nt) {
#pragma unroll
      for (int r = 0; r < 8; ++r) {
        int m = m0 + r + 8 * half;
        int n = d0 + nt * 16 + r15;
        size_t idx = ((size_t)b * S + m) * D + n;
        attn[idx] = o[nt][r] + x[idx];
      }
    }
  }
}

// ---------------------------------------------------------------------------
// Tiled WMMA GEMM: C = epilogue(A[MxK] * B[KxN] + bias)
//   mode 0: GELU -> bf16 out   mode 1: + residual -> f32 out
// 128 threads (4 waves), 64x64 block tile, K chunks of 32.
// A/B tiles staged by TDM (wave 0 issues), double-buffered.
// LDS map (dynamic): At0@0, At1@4096, Bt0@8192, Bt1@12288  (16 KB)
// ---------------------------------------------------------------------------
__global__ __launch_bounds__(128) void k_gemm(
    const unsigned short* __restrict__ A, const unsigned short* __restrict__ Bm,
    const float* __restrict__ bias, const float* __restrict__ resid,
    unsigned short* __restrict__ outb, float* __restrict__ outf,
    int M, int N, int K, int mode) {
  extern __shared__ char gs[];
  int tid  = threadIdx.x;
  int lane = tid & 31;
  int wv   = tid >> 5;
  int nb   = N / 64;
  int m0   = (blockIdx.x / nb) * 64;
  int n0   = (blockIdx.x % nb) * 64;
  int half = lane >> 4;
  int r15  = lane & 15;

  v8f acc[4];
#pragma unroll
  for (int nt = 0; nt < 4; ++nt) { v8f z = {}; acc[nt] = z; }

  int nsteps = K / 32;
  if (wv == 0) {  // prologue: stage k0=0 into buffer 0
    tdm_load_2d_bf16(0u,    A + (size_t)m0 * K, K, 32, 64);
    tdm_load_2d_bf16(8192u, Bm + (size_t)0 * N + n0, N, 64, 32);
  }

  for (int i = 0; i < nsteps; ++i) {
    int k0 = i * 32;
    if (wv == 0) {
      if (i + 1 < nsteps) {
        unsigned int bufn = (unsigned int)((i + 1) & 1) * 4096u;
        tdm_load_2d_bf16(bufn,         A + (size_t)m0 * K + (k0 + 32), K, 32, 64);
        tdm_load_2d_bf16(8192u + bufn, Bm + (size_t)(k0 + 32) * N + n0, N, 64, 32);
        TDM_WAIT(2);
      } else {
        TDM_WAIT(0);
      }
    }
    __syncthreads();
    const unsigned short* At = (const unsigned short*)(gs + (size_t)(i & 1) * 4096);
    const unsigned short* Bt = (const unsigned short*)(gs + 8192 + (size_t)(i & 1) * 4096);

    FragBF a;
#pragma unroll
    for (int v2 = 0; v2 < 8; ++v2) {
      int ka = (v2 >> 2) * 16 + (v2 & 3) * 2 + half * 8;
      a.w[v2] = *(const unsigned int*)&At[(wv * 16 + r15) * 32 + ka];
    }
#pragma unroll
    for (int nt = 0; nt < 4; ++nt) {
      FragBF bf;
#pragma unroll
      for (int v2 = 0; v2 < 8; ++v2) {
        int kb = (v2 >> 2) * 16 + (v2 & 3) * 2 + half * 8;
        bf.u[2 * v2]     = Bt[kb * 64 + nt * 16 + r15];
        bf.u[2 * v2 + 1] = Bt[(kb + 1) * 64 + nt * 16 + r15];
      }
      acc[nt] = __builtin_amdgcn_wmma_f32_16x16x32_bf16(false, a.v, false, bf.v,
                                                        (short)0, acc[nt], false, false);
    }
    __syncthreads();  // all reads done before buffer is overwritten
  }

#pragma unroll
  for (int nt = 0; nt < 4; ++nt) {
#pragma unroll
    for (int r = 0; r < 8; ++r) {
      int m = m0 + wv * 16 + r + 8 * half;
      int n = n0 + nt * 16 + r15;
      float v = acc[nt][r] + bias[n];
      size_t idx = (size_t)m * N + n;
      if (mode == 0) {
        float g = 0.5f * v * (1.0f + erff(v * 0.70710678118654752f));
        outb[idx] = f2bfu(g);
      } else {
        outf[idx] = v + resid[idx];
      }
    }
  }
}

// ---------------------------------------------------------------------------
__global__ void k_f32_to_bf16(const float* __restrict__ s,
                              unsigned short* __restrict__ d, int n) {
  int i = blockIdx.x * blockDim.x + threadIdx.x;
  if (i < n) d[i] = f2bfu(s[i]);
}

// Output projection: VOCAB=33, one block per row
__global__ __launch_bounds__(64) void k_outproj(
    const float* __restrict__ x, const float* __restrict__ w,
    const float* __restrict__ bias, float* __restrict__ out) {
  __shared__ float row[512];
  int rid = blockIdx.x;
  const float* xr = x + (size_t)rid * 512;
  for (int i = threadIdx.x; i < 512; i += 64) row[i] = xr[i];
  __syncthreads();
  if (threadIdx.x < 33) {
    int v = threadIdx.x;
    float s = bias[v];
    for (int d = 0; d < 512; ++d) s += row[d] * w[d * 33 + v];
    out[(size_t)rid * 33 + v] = s;
  }
}

// ---------------------------------------------------------------------------
extern "C" void kernel_launch(void* const* d_in, const int* in_sizes, int n_in,
                              void* d_out, int out_size, void* d_ws, size_t ws_size,
                              hipStream_t stream) {
  constexpr int B = 16, S = 2048, D = 512, L = 4, FF = 2048;
  const int*   ids   = (const int*)d_in[0];
  const int*   cidx  = (const int*)d_in[1];
  const float* amask = (const float*)d_in[2];
  const float* tok   = (const float*)d_in[3];
  const float* pos   = (const float*)d_in[4];
  const float* attr  = (const float*)d_in[5];
  const float* lnw   = (const float*)d_in[6];
  const float* lnb   = (const float*)d_in[7];
  const float* w1    = (const float*)d_in[8];
  const float* b1    = (const float*)d_in[9];
  const float* w2    = (const float*)d_in[10];
  const float* b2    = (const float*)d_in[11];
  const float* outw  = (const float*)d_in[12];
  const float* outbv = (const float*)d_in[13];
  float* logits = (float*)d_out;

  const size_t nTok = (size_t)B * S;   // 32768 rows
  const size_t nXD  = nTok * D;        // 16M elems

  char* p = (char*)d_ws;
  float* x    = (float*)p;              p += nXD * sizeof(float);
  float* attn = (float*)p;              p += nXD * sizeof(float);
  unsigned short* h   = (unsigned short*)p; p += nXD * sizeof(unsigned short);
  unsigned short* y   = (unsigned short*)p; p += nTok * FF * sizeof(unsigned short);
  unsigned short* w1b = (unsigned short*)p; p += (size_t)D * FF * sizeof(unsigned short);
  unsigned short* w2b = (unsigned short*)p;

  k_embed<<<(int)(nXD / 256), 256, 0, stream>>>(ids, cidx, tok, pos, attr, x);

  const size_t smemAttn = 131072 + 128;  // see LDS map above (~128 KB of 320 KB WGP LDS)
  const size_t smemGemm = 16 * 1024;     // 2x(At 4KB + Bt 4KB)

  for (int l = 0; l < L; ++l) {
    k_f32_to_bf16<<<(D * FF) / 256, 256, 0, stream>>>(w1 + (size_t)l * D * FF, w1b, D * FF);
    k_f32_to_bf16<<<(FF * D) / 256, 256, 0, stream>>>(w2 + (size_t)l * FF * D, w2b, FF * D);

    k_layernorm<<<(int)(nTok / 8), 256, 0, stream>>>(x, lnw + l * D, lnb + l * D, h, (int)nTok);
    k_attention<<<B * (S / 16), 32, smemAttn, stream>>>(h, x, amask, attn);
    k_layernorm<<<(int)(nTok / 8), 256, 0, stream>>>(attn, lnw + l * D, lnb + l * D, h, (int)nTok);
    k_gemm<<<(int)((nTok / 64) * (FF / 64)), 128, smemGemm, stream>>>(
        h, w1b, b1 + (size_t)l * FF, nullptr, y, nullptr, (int)nTok, FF, D, 0);
    k_gemm<<<(int)((nTok / 64) * (D / 64)), 128, smemGemm, stream>>>(
        y, w2b, b2 + (size_t)l * D, attn, nullptr, x, (int)nTok, D, FF, 1);
  }
  k_outproj<<<(int)nTok, 64, 0, stream>>>(x, outw, outbv, logits);
}
